// GAT_2138893713777
// MI455X (gfx1250) — matrix operand order repaired and checked by reading, not compiled, not run on hardware
//
#include <hip/hip_runtime.h>

// ---------------------------------------------------------------------------
// GAT 2-layer forward for MI455X (gfx1250, wave32, WMMA).
// Dense GEMMs -> v_wmma_f32_16x16x32_bf16; edge softmax/aggregation -> atomics.
// ---------------------------------------------------------------------------

typedef __attribute__((ext_vector_type(16))) __bf16 v16bf;
typedef __attribute__((ext_vector_type(8)))  __bf16 v8bf;
typedef __attribute__((ext_vector_type(8)))  float  v8f;

#define NNODES 20000
#define NEDGES 320000
#define INF    128
#define HID    32
#define HEADS1 8
#define OUTF   64
#define F1     (HEADS1 * HID)   // 256

// ----- helpers: order-preserving float <-> uint encoding for atomic max -----
__device__ __forceinline__ unsigned encf(float f) {
    unsigned u = __float_as_uint(f);
    return (u & 0x80000000u) ? ~u : (u | 0x80000000u);
}
__device__ __forceinline__ float decf(unsigned u) {
    return (u & 0x80000000u) ? __uint_as_float(u & 0x7FFFFFFFu)
                             : __uint_as_float(~u);
}

// ----- f32 -> bf16 (round to nearest even), stored as raw u16 --------------
__device__ __forceinline__ unsigned short f2bf(float f) {
    unsigned u = __float_as_uint(f);
    u += 0x7FFFu + ((u >> 16) & 1u);
    return (unsigned short)(u >> 16);
}

__global__ void k_cast_bf16(const float* __restrict__ in,
                            unsigned short* __restrict__ out, int n) {
    int i = blockIdx.x * blockDim.x + threadIdx.x;
    if (i < n) out[i] = f2bf(in[i]);
}

// Wt[n*K + k] = W[k*N + n]  (transpose + downconvert)
__global__ void k_transpose_cast_bf16(const float* __restrict__ W,
                                      unsigned short* __restrict__ Wt,
                                      int K, int N) {
    int i = blockIdx.x * blockDim.x + threadIdx.x;
    if (i >= N * K) return;
    int n = i / K, k = i % K;
    Wt[i] = f2bf(W[k * N + n]);
}

// ----- WMMA GEMM: C[M,N] = A[M,K] * B[K,N], A bf16 row-major,
//       B supplied transposed (Bt[N,K] row-major). One wave per 16x16 tile.
__global__ void k_wmma_gemm_bf16(const unsigned short* __restrict__ A,
                                 const unsigned short* __restrict__ Bt,
                                 float* __restrict__ C, int M, int N, int K) {
#if defined(__HIP_DEVICE_COMPILE__)
    int wave  = (int)((blockIdx.x * blockDim.x + threadIdx.x) >> 5);
    int lane  = threadIdx.x & 31;
    int tilesN = N >> 4;
    int tM = (wave / tilesN) << 4;
    int tN = (wave % tilesN) << 4;
    if (tM >= M) return;

    int hv    = lane >> 4;    // lane half: 0 or 1
    int idx16 = lane & 15;    // M index (A) / N index (B)

    // A lane: row tM+idx16, 16 contiguous K starting at hv*16
    const unsigned short* Arow = A  + (size_t)(tM + idx16) * K + (hv << 4);
    // B lane: row (of Bt) tN+idx16, 16 contiguous K starting at hv*16
    const unsigned short* Brow = Bt + (size_t)(tN + idx16) * K + (hv << 4);

    v8f acc = {};
    for (int k0 = 0; k0 < K; k0 += 32) {
        v8bf alo = *(const v8bf*)(Arow + k0);        // K = khalf .. khalf+7
        v8bf ahi = *(const v8bf*)(Arow + k0 + 8);    // K = khalf+8 .. khalf+15
        v16bf a = __builtin_shufflevector(alo, ahi,
                   0,1,2,3,4,5,6,7,8,9,10,11,12,13,14,15);
        v16bf b = *(const v16bf*)(Brow + k0);        // K = khalf .. khalf+15
        acc = __builtin_amdgcn_wmma_f32_16x16x32_bf16(
                  false, a, false, b, (short)0, acc, false, false);
    }

    // C/D layout: VGPR v holds (M = v + hv*8, N = idx16)
    float* Crow = C + (size_t)(tM + (hv << 3)) * N + tN + idx16;
#pragma unroll
    for (int v = 0; v < 8; ++v) Crow[(size_t)v * N] = acc[v];
#endif
}

// ----- per-node attention coefficients: as[n,h] = <H[n,h,:], a_src[h,:]> ----
__global__ void k_attn_coeff(const float* __restrict__ H,
                             const float* __restrict__ a_src,
                             const float* __restrict__ a_dst,
                             float* __restrict__ as_, float* __restrict__ ad_,
                             int nNodes, int heads, int ch) {
    int i = blockIdx.x * blockDim.x + threadIdx.x;
    if (i >= nNodes * heads) return;
    int n = i / heads, h = i % heads;
    const float* hp  = H + (size_t)n * heads * ch + h * ch;
    const float* asp = a_src + h * ch;
    const float* adp = a_dst + h * ch;
    float ss = 0.f, sd = 0.f;
    for (int c = 0; c < ch; ++c) { float v = hp[c]; ss += v * asp[c]; sd += v * adp[c]; }
    as_[i] = ss; ad_[i] = sd;
}

// ----- edge logits (leaky-relu 0.2) + segment max via encoded atomicMax -----
__global__ void k_edge_logits_max(const int* __restrict__ src,
                                  const int* __restrict__ dst,
                                  const float* __restrict__ as_,
                                  const float* __restrict__ ad_,
                                  float* __restrict__ logit,
                                  unsigned* __restrict__ menc,
                                  int nE, int nN, int heads) {
    int e = blockIdx.x * blockDim.x + threadIdx.x;
    int tot = nE + nN;
    if (e >= tot) return;
    int s = (e < nE) ? src[e] : (e - nE);
    int d = (e < nE) ? dst[e] : (e - nE);
    for (int h = 0; h < heads; ++h) {
        float l = as_[s * heads + h] + ad_[d * heads + h];
        l = (l > 0.f) ? l : 0.2f * l;
        logit[(size_t)e * heads + h] = l;
        atomicMax(&menc[d * heads + h], encf(l));
    }
}

// ----- exp(l - max) and segment sum -----------------------------------------
__global__ void k_edge_exp_sum(const int* __restrict__ src,
                               const int* __restrict__ dst,
                               float* __restrict__ logit,
                               const unsigned* __restrict__ menc,
                               float* __restrict__ ssum,
                               int nE, int nN, int heads) {
    int e = blockIdx.x * blockDim.x + threadIdx.x;
    int tot = nE + nN;
    if (e >= tot) return;
    int d = (e < nE) ? dst[e] : (e - nE);
    for (int h = 0; h < heads; ++h) {
        float m  = decf(menc[d * heads + h]);
        float ev = __expf(logit[(size_t)e * heads + h] - m);
        logit[(size_t)e * heads + h] = ev;
        atomicAdd(&ssum[d * heads + h], ev);
    }
}

// ----- message aggregation: out[d,h,:] += alpha * H[s,h,:] ------------------
__global__ void k_edge_aggregate(const int* __restrict__ src,
                                 const int* __restrict__ dst,
                                 const float* __restrict__ H,
                                 const float* __restrict__ eval,
                                 const float* __restrict__ ssum,
                                 float* __restrict__ out,
                                 int nE, int nN, int heads, int ch) {
    int i = blockIdx.x * blockDim.x + threadIdx.x;
    int tot = (nE + nN) * heads;
    if (i >= tot) return;
    int e = i / heads, h = i % heads;
    int s = (e < nE) ? src[e] : (e - nE);
    int d = (e < nE) ? dst[e] : (e - nE);
    float alpha = eval[(size_t)e * heads + h] / (ssum[d * heads + h] + 1e-16f);
    const float* hp = H   + (size_t)s * heads * ch + h * ch;
    float*       op = out + (size_t)d * heads * ch + h * ch;
    for (int c = 0; c < ch; ++c) atomicAdd(&op[c], hp[c] * alpha);
}

// ----- bias add + leaky relu (slope=1 -> identity) --------------------------
__global__ void k_bias_act(float* __restrict__ io, const float* __restrict__ b,
                           int nNodes, int F, float slope) {
    int i = blockIdx.x * blockDim.x + threadIdx.x;
    if (i >= nNodes * F) return;
    int f = i % F;
    float v = io[i] + b[f];
    io[i] = (v > 0.f) ? v : slope * v;
}

// ---------------------------------------------------------------------------
extern "C" void kernel_launch(void* const* d_in, const int* in_sizes, int n_in,
                              void* d_out, int out_size, void* d_ws, size_t ws_size,
                              hipStream_t stream) {
    (void)in_sizes; (void)n_in; (void)ws_size;
    const float* x      = (const float*)d_in[0];
    const float* W1     = (const float*)d_in[1];
    const float* a_src1 = (const float*)d_in[2];
    const float* a_dst1 = (const float*)d_in[3];
    const float* b1     = (const float*)d_in[4];
    const float* W2     = (const float*)d_in[5];
    const float* a_src2 = (const float*)d_in[6];
    const float* a_dst2 = (const float*)d_in[7];
    const float* b2     = (const float*)d_in[8];
    const int*   ei     = (const int*)d_in[9];
    const int* srcI = ei;
    const int* dstI = ei + NEDGES;
    float* out = (float*)d_out;

    // ---- workspace carve-out (256B aligned) ----
    char* base = (char*)d_ws;
    size_t off = 0;
    auto alloc = [&](size_t bytes) -> void* {
        void* p = base + off;
        off += (bytes + 255) & ~(size_t)255;
        return p;
    };
    unsigned short* xb    = (unsigned short*)alloc((size_t)NNODES * INF * 2);  // x bf16
    unsigned short* w1t   = (unsigned short*)alloc((size_t)F1 * INF * 2);      // W1^T bf16
    float*          h1    = (float*)alloc((size_t)NNODES * F1 * 4);            // x@W1
    float*          as1   = (float*)alloc((size_t)NNODES * HEADS1 * 4);
    float*          ad1   = (float*)alloc((size_t)NNODES * HEADS1 * 4);
    unsigned*       m1    = (unsigned*)alloc((size_t)NNODES * HEADS1 * 4);
    float*          s1    = (float*)alloc((size_t)NNODES * HEADS1 * 4);
    float*          lg1   = (float*)alloc((size_t)(NEDGES + NNODES) * HEADS1 * 4);
    float*          agg1  = (float*)alloc((size_t)NNODES * F1 * 4);            // layer1 out / hact
    unsigned short* hactb = (unsigned short*)alloc((size_t)NNODES * F1 * 2);
    unsigned short* w2t   = (unsigned short*)alloc((size_t)OUTF * F1 * 2);
    float*          t2    = (float*)alloc((size_t)NNODES * OUTF * 4);          // hact@W2
    float*          as2   = (float*)alloc((size_t)NNODES * 4);
    float*          ad2   = (float*)alloc((size_t)NNODES * 4);
    unsigned*       m2    = (unsigned*)alloc((size_t)NNODES * 4);
    float*          s2    = (float*)alloc((size_t)NNODES * 4);
    float*          lg2   = (float*)alloc((size_t)(NEDGES + NNODES) * 4);

    const int ET = NEDGES + NNODES;  // edges incl. self loops
    auto grid = [](int n) { return (n + 255) / 256; };

    // ---- zero-init accumulators (graph-capture-safe async memsets) ----
    hipMemsetAsync(m1,   0, (size_t)NNODES * HEADS1 * 4, stream);
    hipMemsetAsync(s1,   0, (size_t)NNODES * HEADS1 * 4, stream);
    hipMemsetAsync(agg1, 0, (size_t)NNODES * F1 * 4,     stream);
    hipMemsetAsync(m2,   0, (size_t)NNODES * 4,          stream);
    hipMemsetAsync(s2,   0, (size_t)NNODES * 4,          stream);
    hipMemsetAsync(out,  0, (size_t)out_size * 4,        stream);

    // ============== Layer 1 ==============
    k_cast_bf16<<<grid(NNODES * INF), 256, 0, stream>>>(x, xb, NNODES * INF);
    k_transpose_cast_bf16<<<grid(F1 * INF), 256, 0, stream>>>(W1, w1t, INF, F1);
    // 1250 x 16 tiles = 20000 waves, 8 waves/block
    k_wmma_gemm_bf16<<<(NNODES / 16) * (F1 / 16) / 8, 256, 0, stream>>>(
        xb, w1t, h1, NNODES, F1, INF);
    k_attn_coeff<<<grid(NNODES * HEADS1), 256, 0, stream>>>(
        h1, a_src1, a_dst1, as1, ad1, NNODES, HEADS1, HID);
    k_edge_logits_max<<<grid(ET), 256, 0, stream>>>(
        srcI, dstI, as1, ad1, lg1, m1, NEDGES, NNODES, HEADS1);
    k_edge_exp_sum<<<grid(ET), 256, 0, stream>>>(
        srcI, dstI, lg1, m1, s1, NEDGES, NNODES, HEADS1);
    k_edge_aggregate<<<grid(ET * HEADS1), 256, 0, stream>>>(
        srcI, dstI, h1, lg1, s1, agg1, NEDGES, NNODES, HEADS1, HID);
    k_bias_act<<<grid(NNODES * F1), 256, 0, stream>>>(agg1, b1, NNODES, F1, 0.01f);

    // ============== Layer 2 ==============
    k_cast_bf16<<<grid(NNODES * F1), 256, 0, stream>>>(agg1, hactb, NNODES * F1);
    k_transpose_cast_bf16<<<grid(OUTF * F1), 256, 0, stream>>>(W2, w2t, F1, OUTF);
    // 1250 x 4 tiles = 5000 waves, 8 waves/block
    k_wmma_gemm_bf16<<<(NNODES / 16) * (OUTF / 16) / 8, 256, 0, stream>>>(
        hactb, w2t, t2, NNODES, OUTF, F1);
    k_attn_coeff<<<grid(NNODES), 256, 0, stream>>>(
        t2, a_src2, a_dst2, as2, ad2, NNODES, 1, OUTF);
    k_edge_logits_max<<<grid(ET), 256, 0, stream>>>(
        srcI, dstI, as2, ad2, lg2, m2, NEDGES, NNODES, 1);
    k_edge_exp_sum<<<grid(ET), 256, 0, stream>>>(
        srcI, dstI, lg2, m2, s2, NEDGES, NNODES, 1);
    k_edge_aggregate<<<grid(ET), 256, 0, stream>>>(
        srcI, dstI, t2, lg2, s2, out, NEDGES, NNODES, 1, OUTF);
    k_bias_act<<<grid(NNODES * OUTF), 256, 0, stream>>>(out, b2, NNODES, OUTF, 1.0f);
}
